// RNN_36575941492858
// MI455X (gfx1250) — matrix operand order repaired and checked by reading, not compiled, not run on hardware
//
#include <hip/hip_runtime.h>
#include <hip/hip_bf16.h>

typedef __attribute__((ext_vector_type(16))) _Float16 v16h;
typedef __attribute__((ext_vector_type(8)))  _Float16 v8h;
typedef __attribute__((ext_vector_type(8)))  float    v8f;
typedef __attribute__((ext_vector_type(4)))  float    v4f;

#define T_STEPS   512
#define BATCH     64
#define RNN_DIM   512
#define IN_DIM    512
#define LDS_K     520   // 512 + 8 halfs pad (16B) -> row stride 1040B, breaks bank conflicts

// Branch-free tanh: e2 = exp(-2|x|) in (0,1]; tanh(|x|) = (1-e2)/(1+e2).
__device__ __forceinline__ float fast_tanh(float x) {
    float e2 = __expf(-2.0f * __builtin_fabsf(x));
    float r  = (1.0f - e2) / (1.0f + e2);
    return __builtin_copysignf(r, x);
}

// ---------------------------------------------------------------------------
// Phase 1: XWb[t*64+b, n] = sum_k X[t,b,k] * W[k,n] + bias[n]   (written to d_out)
// Fully parallel GEMM: M=32768, K=512, N=512. f16 WMMA, f32 accumulate.
// grid = (16 col-groups of 32 cols, 256 row-groups of 128 rows), block = 256 (8 waves).
// ---------------------------------------------------------------------------
__global__ __launch_bounds__(256)
void rnn_xw_kernel(const float* __restrict__ X,     // [T*B, 512]
                   const float* __restrict__ W,     // [1024, 512]
                   const float* __restrict__ bias,  // [512]
                   float* __restrict__ out)         // [T*B, 512]
{
    __shared__ __align__(16) _Float16 Wl[32 * LDS_K];  // [colLocal][k], column-major slice

    const int tid     = threadIdx.x;
    const int colbase = blockIdx.x * 32;
    const int rbase   = blockIdx.y * 128;

    // Stage W_x[:, colbase:colbase+32] into LDS as f16, column-major.
    for (int idx = tid; idx < 32 * 512; idx += 256) {
        int k  = idx >> 5;   // 0..511
        int cl = idx & 31;   // 0..31
        Wl[cl * LDS_K + k] = (_Float16)W[(size_t)k * 512 + colbase + cl];
    }
    __syncthreads();

    const int wave  = tid >> 5;
    const int lane  = tid & 31;
    const int nhat  = lane & 15;
    const int khalf = lane >> 4;

    const float* rowp = X + (size_t)(rbase + wave * 16 + nhat) * 512;  // A row = lane&15

    v8f acc[2];
    {
        float b0 = bias[colbase + nhat];
        float b1 = bias[colbase + 16 + nhat];
        #pragma unroll
        for (int i = 0; i < 8; ++i) { acc[0][i] = b0; acc[1][i] = b1; }
    }

    #pragma unroll 4
    for (int kf = 0; kf < 16; ++kf) {
        // A fragment (16x32 f16): lane m -> K kf*32+khalf*8..+7 and +16..+23
        int k0 = kf * 32 + khalf * 8;
        v4f xa = *(const v4f*)(rowp + k0);
        v4f xb = *(const v4f*)(rowp + k0 + 4);
        v4f xc = *(const v4f*)(rowp + k0 + 16);
        v4f xd = *(const v4f*)(rowp + k0 + 20);
        v16h a;
        #pragma unroll
        for (int j = 0; j < 4; ++j) {
            a[j]      = (_Float16)xa[j];
            a[4 + j]  = (_Float16)xb[j];
            a[8 + j]  = (_Float16)xc[j];
            a[12 + j] = (_Float16)xd[j];
        }
        #pragma unroll
        for (int c = 0; c < 2; ++c) {
            // B fragment (32x16 f16): lane n -> col n, K kf*32+khalf*8..+7 and +16..+23
            const _Float16* bp = &Wl[(c * 16 + nhat) * LDS_K + kf * 32 + khalf * 8];
            v8h blo = *(const v8h*)bp;
            v8h bhi = *(const v8h*)(bp + 16);
            v16h b;
            #pragma unroll
            for (int j = 0; j < 8; ++j) { b[j] = blo[j]; b[8 + j] = bhi[j]; }
            acc[c] = __builtin_amdgcn_wmma_f32_16x16x32_f16(
                false, a, false, b, (short)0, acc[c], false, false);
        }
    }

    // C/D layout: VGPR i -> (M = i + 8*khalf, N = nhat)
    #pragma unroll
    for (int c = 0; c < 2; ++c)
        #pragma unroll
        for (int i = 0; i < 8; ++i)
            out[(size_t)(rbase + wave * 16 + i + 8 * khalf) * 512 + colbase + c * 16 + nhat]
                = acc[c][i];
}

// ---------------------------------------------------------------------------
// Phase 2: persistent recurrence. 4 workgroups (one per 16 batch rows), 512
// threads = 16 waves. Wave w owns column tiles {2w, 2w+1} of W_h, resident as
// 2x16 v16h B-fragments (256 VGPRs). h double-buffered in LDS as f16; exact
// f32 h carried in registers for the output. xwb loads are software-pipelined
// one step ahead so the per-step critical path is: barrier -> ds h loads ->
// 2x16 chained WMMA -> add xw -> tanh -> ds store -> barrier.
// d_out[t] holds xwb on entry; overwritten with PRE-update hidden (== h_t).
// ---------------------------------------------------------------------------
__global__ __launch_bounds__(512)
void rnn_recur_kernel(const float* __restrict__ W,      // [1024, 512]; W_h = rows 512..1023
                      const float* __restrict__ init_h, // [512]
                      float* __restrict__ out)          // [T, 64, 512]
{
    __shared__ __align__(16) _Float16 hbuf[2][16 * LDS_K];  // [buf][m][k]

    const int tid   = threadIdx.x;
    const int wave  = tid >> 5;          // 0..15 -> column tiles 2w, 2w+1
    const int lane  = tid & 31;
    const int nhat  = lane & 15;
    const int khalf = lane >> 4;
    const int col0  = wave * 32 + nhat;  // tile c column = col0 + c*16
    const int bbase = blockIdx.x * 16;

    // Resident W_h fragments for this wave's two column tiles (one-time gather).
    v16h bfrag[2][16];
    #pragma unroll
    for (int c = 0; c < 2; ++c)
        #pragma unroll
        for (int kf = 0; kf < 16; ++kf) {
            int kb = 512 + kf * 32 + khalf * 8;  // W row (K index + W_h offset)
            #pragma unroll
            for (int j = 0; j < 16; ++j)
                bfrag[c][kf][j] = (_Float16)W[(size_t)(kb + (j < 8 ? j : j + 8)) * 512 + col0 + c * 16];
        }

    // h_0 = broadcast(init_hidden); exact f32 carry + f16 matmul copy in LDS.
    float hprev[2][8];
    #pragma unroll
    for (int c = 0; c < 2; ++c) {
        float h0 = init_h[col0 + c * 16];
        #pragma unroll
        for (int i = 0; i < 8; ++i) {
            hprev[c][i] = h0;
            hbuf[0][(i + 8 * khalf) * LDS_K + col0 + c * 16] = (_Float16)h0;
        }
    }

    // Prefetch xwb for t = 0.
    float xw[2][8];
    {
        const float* p = out + (size_t)bbase * 512;
        #pragma unroll
        for (int c = 0; c < 2; ++c)
            #pragma unroll
            for (int i = 0; i < 8; ++i)
                xw[c][i] = p[(size_t)(i + 8 * khalf) * 512 + col0 + c * 16];
    }
    __syncthreads();

    for (int t = 0; t < T_STEPS; ++t) {
        float* outp = out + ((size_t)t * BATCH + bbase) * 512;

        // acc = h_t @ W_h[:, tiles]  (2 independent chains of 16 WMMAs, K=512)
        v8f acc[2];
        #pragma unroll
        for (int c = 0; c < 2; ++c)
            #pragma unroll
            for (int i = 0; i < 8; ++i) acc[c][i] = 0.0f;

        const _Float16* hb = hbuf[t & 1];
        #pragma unroll
        for (int kf = 0; kf < 16; ++kf) {
            const _Float16* ap = hb + nhat * LDS_K + kf * 32 + khalf * 8;
            v8h alo = *(const v8h*)ap;         // K kf*32+khalf*8 .. +7
            v8h ahi = *(const v8h*)(ap + 16);  // K kf*32+khalf*8+16 .. +23
            v16h a;
            #pragma unroll
            for (int j = 0; j < 8; ++j) { a[j] = alo[j]; a[8 + j] = ahi[j]; }
            acc[0] = __builtin_amdgcn_wmma_f32_16x16x32_f16(
                false, a, false, bfrag[0][kf], (short)0, acc[0], false, false);
            acc[1] = __builtin_amdgcn_wmma_f32_16x16x32_f16(
                false, a, false, bfrag[1][kf], (short)0, acc[1], false, false);
        }

        // acc += xwb[t] (prefetched; use forces the loadcnt wait here, off the
        // top of the chain), then overwrite out[t] with the PRE-update hidden.
        #pragma unroll
        for (int c = 0; c < 2; ++c)
            #pragma unroll
            for (int i = 0; i < 8; ++i) acc[c][i] += xw[c][i];
        asm volatile("s_wait_loadcnt 0x0" ::: "memory");  // safety: reads done before overwrite

        #pragma unroll
        for (int c = 0; c < 2; ++c)
            #pragma unroll
            for (int i = 0; i < 8; ++i)
                outp[(size_t)(i + 8 * khalf) * 512 + col0 + c * 16] = hprev[c][i];

        // Prefetch next step's xwb (overlaps tanh + barrier + next wmma chain).
        if (t + 1 < T_STEPS) {
            const float* pn = outp + (size_t)BATCH * 512;
            #pragma unroll
            for (int c = 0; c < 2; ++c)
                #pragma unroll
                for (int i = 0; i < 8; ++i)
                    xw[c][i] = pn[(size_t)(i + 8 * khalf) * 512 + col0 + c * 16];
        }

        // h_{t+1} = tanh(acc): exact f32 carry + f16 staging for next A frags.
        _Float16* hn = hbuf[(t + 1) & 1];
        #pragma unroll
        for (int c = 0; c < 2; ++c)
            #pragma unroll
            for (int i = 0; i < 8; ++i) {
                float v = fast_tanh(acc[c][i]);
                hprev[c][i] = v;
                hn[(i + 8 * khalf) * LDS_K + col0 + c * 16] = (_Float16)v;
            }
        __syncthreads();  // next buffer complete; old buffer free for t+2
    }
}

// ---------------------------------------------------------------------------
extern "C" void kernel_launch(void* const* d_in, const int* in_sizes, int n_in,
                              void* d_out, int out_size, void* d_ws, size_t ws_size,
                              hipStream_t stream) {
    const float* X    = (const float*)d_in[0];  // [512,64,512] f32
    const float* W    = (const float*)d_in[1];  // [1024,512]   f32
    const float* bias = (const float*)d_in[2];  // [1,512]      f32
    const float* h0   = (const float*)d_in[3];  // [1,512]      f32
    float* out = (float*)d_out;                 // [512,64,512] f32

    // Phase 1: xwb -> d_out. 16 col-groups x 256 row-groups, 256 threads.
    rnn_xw_kernel<<<dim3(16, 256), dim3(256), 0, stream>>>(X, W, bias, out);
    // Phase 2: persistent recurrence, one workgroup per 16 batch rows.
    rnn_recur_kernel<<<dim3(4), dim3(512), 0, stream>>>(W, h0, out);
}
